// Mamba2Enc_2284922601743
// MI455X (gfx1250) — matrix operand order repaired and checked by reading, not compiled, not run on hardware
//
#include <hip/hip_runtime.h>
#include <hip/hip_bf16.h>
#include <math.h>

typedef __attribute__((ext_vector_type(16))) __bf16 v16bf;
typedef __attribute__((ext_vector_type(8)))  float  v8f;

#define D_IN      1024
#define D_MODEL   512
#define NLAYERS   2
#define D_STATE   64
#define D_CONV    4
#define HEADDIM   64
#define D_INNER   1024
#define NHEADS    16
#define CONV_DIM  1152   // D_INNER + 2*D_STATE
#define D_IN_PROJ 2192   // 2*D_INNER + 2*D_STATE + NHEADS
#define SEQ       4096
#define NBATCH    4
#define NTOK      (NBATCH*SEQ)
#define EPSV      1e-5f
#define QL        64               // chunk length for SSD scan
#define NCHUNK    (SEQ/QL)         // 64
#define LDB       66               // padded bf16 LDS stride

__device__ __forceinline__ float sigmoidf_(float x){ return 1.f/(1.f+expf(-x)); }
__device__ __forceinline__ float geluf_(float x){ return 0.5f*x*(1.f+erff(x*0.70710678118654752f)); }

// ---- WMMA fragment gathers (CDNA5 wave32 layouts, bf16 16x16x32) ----------
// A (16x32): lane&15 = row; lanes>=16 use K+8; VGPR v -> K pairs
__device__ __forceinline__ v16bf afrag_load(const __bf16* m, int ld, int row0, int k0, int lane){
  const __bf16* p = m + (size_t)(row0 + (lane & 15)) * ld;
  const int kb = k0 + ((lane >> 4) << 3);
  v16bf f;
#pragma unroll
  for (int v = 0; v < 8; ++v){
    const int k = kb + ((v & 4) ? 16 : 0) + (v & 3) * 2;
    f[2*v] = p[k]; f[2*v+1] = p[k+1];
  }
  return f;
}
// B (32x16): lane&15 = column; element [k][col] gathered from storage m[col][k]
__device__ __forceinline__ v16bf bfrag_load(const __bf16* m, int ld, int col0, int k0, int lane){
  const __bf16* p = m + (size_t)(col0 + (lane & 15)) * ld;
  const int kb = k0 + ((lane >> 4) << 3);
  v16bf f;
#pragma unroll
  for (int v = 0; v < 8; ++v){ const int k = kb + ((v & 4) ? 16 : 0) + (v & 3) * 2; f[2*v] = p[k]; f[2*v+1] = p[k + 1]; }
  return f;
}

// ---- async global->LDS helpers (ASYNCcnt path) ----------------------------
__device__ __forceinline__ void async_ld_b32(void* lds_ptr, const void* gptr){
  const unsigned lds_off = (unsigned)(size_t)lds_ptr;         // flat low 32 = LDS addr
  asm volatile("global_load_async_to_lds_b32 %0, %1, off"
               :: "v"(lds_off), "v"((unsigned long long)(size_t)gptr) : "memory");
}
__device__ __forceinline__ void wait_async0(){
  asm volatile("s_wait_asynccnt 0x0" ::: "memory");
}

// ---------------------------------------------------------------------------
// bf16 WMMA GEMM:  C[m][n] = epi( sum_k A[m][k]*W[n][k] )   (W is N x K)
// epi: +bias[n], optional exact GELU, optional +residual[m][n]
// Requires M % 128 == 0 and K % 32 == 0 (true for all calls); N arbitrary.
// ---------------------------------------------------------------------------
#define TM 128
#define TN 128
#define TKS 32
#define LDST (TKS+2)

__global__ __launch_bounds__(256) void gemm_bf16_wmma(
    const float* __restrict__ A, const float* __restrict__ W,
    const float* __restrict__ bias, const float* __restrict__ residual,
    float* __restrict__ C, int M, int N, int K, int act)
{
  __shared__ __bf16 lA[TM][LDST];
  __shared__ __bf16 lB[TN][LDST];
  const int tid  = threadIdx.x;
  const int wave = tid >> 5, lane = tid & 31;
  const int bm = blockIdx.x * TM, bn = blockIdx.y * TN;
  const int wm = (wave >> 1) * 32, wn = (wave & 1) * 64;

  v8f acc[2][4];
#pragma unroll
  for (int i = 0; i < 2; ++i)
#pragma unroll
    for (int j = 0; j < 4; ++j)
#pragma unroll
      for (int v = 0; v < 8; ++v) acc[i][j][v] = 0.f;

  const int lr = tid >> 1;
  const int lc = (tid & 1) * 16;
  const float* aptr = A + (size_t)(bm + lr) * K + lc;
  const int    gn   = bn + lr;
  const float* wptr = (gn < N) ? (W + (size_t)gn * K + lc) : nullptr;

  for (int k0 = 0; k0 < K; k0 += TKS) {
#pragma unroll
    for (int j = 0; j < 16; ++j) lA[lr][lc + j] = (__bf16)aptr[k0 + j];
    if (wptr) {
#pragma unroll
      for (int j = 0; j < 16; ++j) lB[lr][lc + j] = (__bf16)wptr[k0 + j];
    } else {
#pragma unroll
      for (int j = 0; j < 16; ++j) lB[lr][lc + j] = (__bf16)0.f;
    }
    if (k0 + TKS < K) {                     // global_prefetch_b8
      __builtin_prefetch(aptr + k0 + TKS, 0, 0);
      if (wptr) __builtin_prefetch(wptr + k0 + TKS, 0, 0);
    }
    __syncthreads();

    v16bf afrag[2];
#pragma unroll
    for (int i = 0; i < 2; ++i)
      afrag[i] = afrag_load(&lA[0][0], LDST, wm + i * 16, 0, lane);
    v16bf bfrag[4];
#pragma unroll
    for (int j = 0; j < 4; ++j)
      bfrag[j] = bfrag_load(&lB[0][0], LDST, wn + j * 16, 0, lane);
#pragma unroll
    for (int i = 0; i < 2; ++i)
#pragma unroll
      for (int j = 0; j < 4; ++j)
        acc[i][j] = __builtin_amdgcn_wmma_f32_16x16x32_bf16(
            false, afrag[i], false, bfrag[j], (short)0, acc[i][j], false, false);
    __syncthreads();
  }

  const int cn  = lane & 15;
  const int cmb = (lane >> 4) * 8;
#pragma unroll
  for (int i = 0; i < 2; ++i)
#pragma unroll
    for (int j = 0; j < 4; ++j) {
      const int gcol = bn + wn + j * 16 + cn;
      if (gcol < N) {
#pragma unroll
        for (int v = 0; v < 8; ++v) {
          const int grow = bm + wm + i * 16 + cmb + v;
          float r = acc[i][j][v];
          if (bias)     r += bias[gcol];
          if (act == 1) r  = geluf_(r);
          if (residual) r += residual[(size_t)grow * N + gcol];
          C[(size_t)grow * N + gcol] = r;
        }
      }
    }
}

// ---------------------------------------------------------------------------
// LayerNorm over D_MODEL=512 (one block / token)
// ---------------------------------------------------------------------------
__global__ __launch_bounds__(256) void layernorm_kernel(
    const float* __restrict__ h, const float* __restrict__ w,
    const float* __restrict__ b, float* __restrict__ out)
{
  const int t = blockIdx.x, tid = threadIdx.x;
  const float* row = h + (size_t)t * D_MODEL;
  const float v0 = row[tid], v1 = row[tid + 256];
  __shared__ float red[256];
  red[tid] = v0 + v1; __syncthreads();
  for (int s = 128; s > 0; s >>= 1) { if (tid < s) red[tid] += red[tid + s]; __syncthreads(); }
  const float mu = red[0] * (1.f / 512.f);
  __syncthreads();
  const float d0 = v0 - mu, d1 = v1 - mu;
  red[tid] = d0 * d0 + d1 * d1; __syncthreads();
  for (int s = 128; s > 0; s >>= 1) { if (tid < s) red[tid] += red[tid + s]; __syncthreads(); }
  const float rs = rsqrtf(red[0] * (1.f / 512.f) + EPSV);
  float* orow = out + (size_t)t * D_MODEL;
  orow[tid]       = d0 * rs * w[tid]       + b[tid];
  orow[tid + 256] = d1 * rs * w[tid + 256] + b[tid + 256];
}

// ---------------------------------------------------------------------------
// Causal depthwise conv (width 4) + SiLU, plus dt = softplus(raw + dt_bias).
// ---------------------------------------------------------------------------
__global__ __launch_bounds__(256) void convdt_kernel(
    const float* __restrict__ zx, const float* __restrict__ cw,
    const float* __restrict__ cb, const float* __restrict__ dtbias,
    float* __restrict__ conv, float* __restrict__ dtb)
{
  const int t = blockIdx.x;
  const int l = t & (SEQ - 1);
  const float* rbase = zx + (size_t)t * D_IN_PROJ + D_INNER;
  for (int c = threadIdx.x; c < CONV_DIM; c += 256) {
    float acc = cb[c];
#pragma unroll
    for (int k = 0; k < D_CONV; ++k) {
      const int dl = l - (D_CONV - 1) + k;
      if (dl >= 0)
        acc += cw[c * D_CONV + k] * rbase[(ptrdiff_t)(k - (D_CONV - 1)) * D_IN_PROJ + c];
    }
    acc = acc * sigmoidf_(acc);               // SiLU
    conv[(size_t)t * CONV_DIM + c] = acc;
  }
  if (threadIdx.x < NHEADS) {
    const int hh = threadIdx.x;
    const float r  = zx[(size_t)t * D_IN_PROJ + (D_IN_PROJ - NHEADS) + hh] + dtbias[hh];
    dtb[(size_t)t * NHEADS + hh] = (r > 20.f) ? r : log1pf(expf(r));
  }
}

// ---------------------------------------------------------------------------
// SSD pass 1: per (b, head, chunk) tile.
//   G = C.B^T masked by exp(s_t - s_s) (causal), Y_local = G.(dt*X) + D*x,
//   dH = (B * gamma)^T.(dt*X),  with s = A*cumsum(dt), gamma_s = exp(s_63-s_s).
// dt staged via global_load_async_to_lds_b32 (ASYNCcnt path).
// ---------------------------------------------------------------------------
__global__ __launch_bounds__(256) void ssd_pass1(
    const float* __restrict__ conv, const float* __restrict__ dtb,
    const float* __restrict__ Alog, const float* __restrict__ Dsk,
    float* __restrict__ y, float* __restrict__ Hbuf,
    float* __restrict__ dAc, float* __restrict__ scumb)
{
  const int blk = blockIdx.x;
  const int c  = blk & (NCHUNK - 1);
  const int bh = blk >> 6;                 // b*16 + head
  const int b  = bh >> 4, hh = bh & 15;
  const int tid = threadIdx.x, wave = tid >> 5, lane = tid & 31;

  __shared__ float sdt[QL];
  __shared__ float scum[QL];
  __shared__ __bf16 lB [QL][LDB];          // [s][n]
  __shared__ __bf16 lC [QL][LDB];          // [t][n]
  __shared__ __bf16 lXT[QL][LDB];          // [p][s]  (dt-scaled X, transposed)
  __shared__ __bf16 lG [QL][LDB];          // [t][s]  masked C.B^T
  __shared__ __bf16 lBg[QL][LDB];          // [n][s]  (B*gamma)^T

  const float* cbase = conv + ((size_t)b * SEQ + (size_t)c * QL) * CONV_DIM;
  const float* dtrow = dtb  + ((size_t)b * SEQ + (size_t)c * QL) * NHEADS + hh;

  // async-gather this chunk's 64 dt values into LDS
  if (tid < QL)
    async_ld_b32(&sdt[tid], dtrow + (size_t)tid * NHEADS);

  // stage B and C tiles (bf16) while the async copy is in flight
  {
    const int r = tid >> 2, c0 = (tid & 3) * 16;
    const float* rowp = cbase + (size_t)r * CONV_DIM;
#pragma unroll
    for (int j = 0; j < 16; ++j) lB[r][c0 + j] = (__bf16)rowp[D_INNER + c0 + j];
#pragma unroll
    for (int j = 0; j < 16; ++j) lC[r][c0 + j] = (__bf16)rowp[D_INNER + D_STATE + c0 + j];
  }
  wait_async0();
  __syncthreads();

  if (tid == 0) {                          // decay cumsum: s_t = A * cumsum(dt)
    const float Av = -expf(Alog[hh]);
    float run = 0.f;
    for (int t = 0; t < QL; ++t) { run += sdt[t]; scum[t] = run * Av; }
  }
  __syncthreads();

  // stage dt-scaled X transposed; export per-chunk decay state
  {
    const int s = tid >> 2, p0 = (tid & 3) * 16;
    const float d = sdt[s];
    const float* rowp = cbase + (size_t)s * CONV_DIM + hh * HEADDIM;
#pragma unroll
    for (int j = 0; j < 16; ++j) lXT[p0 + j][s] = (__bf16)(rowp[p0 + j] * d);
  }
  if (tid < QL) scumb[(size_t)bh * SEQ + (size_t)c * QL + tid] = scum[tid];
  if (tid == 0) dAc[(size_t)bh * NCHUNK + c] = expf(scum[QL - 1]);
  __syncthreads();

  const float stot = scum[QL - 1];
  const int cn  = lane & 15;
  const int cmb = (lane >> 4) * 8;

  // G = C.B^T (64x64, K=64), masked + decayed, written back as bf16
#pragma unroll
  for (int rep = 0; rep < 2; ++rep) {
    const int ti = wave + rep * 8;
    const int tr = (ti >> 2) * 16, tc = (ti & 3) * 16;
    v8f acc;
#pragma unroll
    for (int v = 0; v < 8; ++v) acc[v] = 0.f;
#pragma unroll
    for (int kk = 0; kk < 2; ++kk)
      acc = __builtin_amdgcn_wmma_f32_16x16x32_bf16(
          false, afrag_load(&lC[0][0], LDB, tr, kk * 32, lane),
          false, bfrag_load(&lB[0][0], LDB, tc, kk * 32, lane),
          (short)0, acc, false, false);
#pragma unroll
    for (int v = 0; v < 8; ++v) {
      const int t = tr + cmb + v, s = tc + cn;
      lG[t][s] = (__bf16)((t >= s) ? acc[v] * expf(scum[t] - scum[s]) : 0.f);
    }
  }
  // (B*gamma)^T  [n][s]
  {
    const int n = tid >> 2, s0 = (tid & 3) * 16;
#pragma unroll
    for (int j = 0; j < 16; ++j) {
      const int s = s0 + j;
      lBg[n][s] = (__bf16)((float)lB[s][n] * expf(stot - scum[s]));
    }
  }
  __syncthreads();

  const float dskip = Dsk[hh];
  float* ybase = y + ((size_t)b * SEQ + (size_t)c * QL) * D_INNER + hh * HEADDIM;
  float* hbase = Hbuf + ((size_t)bh * NCHUNK + c) * (D_STATE * HEADDIM);

#pragma unroll
  for (int rep = 0; rep < 2; ++rep) {
    const int ti = wave + rep * 8;
    const int tr = (ti >> 2) * 16, tc = (ti & 3) * 16;
    // Y_local = G.(dt*X)
    v8f acc;
#pragma unroll
    for (int v = 0; v < 8; ++v) acc[v] = 0.f;
#pragma unroll
    for (int kk = 0; kk < 2; ++kk)
      acc = __builtin_amdgcn_wmma_f32_16x16x32_bf16(
          false, afrag_load(&lG[0][0], LDB, tr, kk * 32, lane),
          false, bfrag_load(&lXT[0][0], LDB, tc, kk * 32, lane),
          (short)0, acc, false, false);
#pragma unroll
    for (int v = 0; v < 8; ++v) {
      const int t = tr + cmb + v, p = tc + cn;
      const float xv = cbase[(size_t)t * CONV_DIM + hh * HEADDIM + p];
      ybase[(size_t)t * D_INNER + p] = acc[v] + dskip * xv;
    }
    // dH = (B*gamma)^T.(dt*X)
    v8f hcc;
#pragma unroll
    for (int v = 0; v < 8; ++v) hcc[v] = 0.f;
#pragma unroll
    for (int kk = 0; kk < 2; ++kk)
      hcc = __builtin_amdgcn_wmma_f32_16x16x32_bf16(
          false, afrag_load(&lBg[0][0], LDB, tr, kk * 32, lane),
          false, bfrag_load(&lXT[0][0], LDB, tc, kk * 32, lane),
          (short)0, hcc, false, false);
#pragma unroll
    for (int v = 0; v < 8; ++v)
      hbase[(size_t)(tr + cmb + v) * HEADDIM + (tc + cn)] = hcc[v];
  }
}

// ---------------------------------------------------------------------------
// SSD pass 2: chunk-level recurrence. Hbuf[c] (= dH) is replaced in place by
// H_in[c]; H <- exp(s_chunk)*H + dH. 64 blocks, state in registers.
// ---------------------------------------------------------------------------
__global__ __launch_bounds__(256) void ssd_pass2(
    float* __restrict__ Hbuf, const float* __restrict__ dAc)
{
  const int bh = blockIdx.x, tid = threadIdx.x;
  float Hreg[16];
#pragma unroll
  for (int j = 0; j < 16; ++j) Hreg[j] = 0.f;
  for (int c = 0; c < NCHUNK; ++c) {
    float* p = Hbuf + ((size_t)bh * NCHUNK + c) * (D_STATE * HEADDIM);
    const float a = dAc[(size_t)bh * NCHUNK + c];
#pragma unroll
    for (int j = 0; j < 16; ++j) {
      const int e = tid + j * 256;
      const float tmp = p[e];
      p[e] = Hreg[j];
      Hreg[j] = Hreg[j] * a + tmp;
    }
  }
}

// ---------------------------------------------------------------------------
// SSD pass 3: Y += (Lambda * C) . H_in, per (b, head, chunk) tile.
// ---------------------------------------------------------------------------
__global__ __launch_bounds__(256) void ssd_pass3(
    const float* __restrict__ Hbuf, const float* __restrict__ conv,
    const float* __restrict__ scumb, float* __restrict__ y)
{
  const int blk = blockIdx.x;
  const int c  = blk & (NCHUNK - 1);
  const int bh = blk >> 6;
  const int b  = bh >> 4, hh = bh & 15;
  const int tid = threadIdx.x, wave = tid >> 5, lane = tid & 31;

  __shared__ __bf16 lCl[QL][LDB];          // [t][n] = exp(s_t) * C[t][n]
  __shared__ __bf16 lHT[QL][LDB];          // [p][n]

  const float* cbase = conv + ((size_t)b * SEQ + (size_t)c * QL) * CONV_DIM;
  const float* hbase = Hbuf + ((size_t)bh * NCHUNK + c) * (D_STATE * HEADDIM);
  {
    const int t = tid >> 2, n0 = (tid & 3) * 16;
    const float lam = expf(scumb[(size_t)bh * SEQ + (size_t)c * QL + t]);
    const float* rowp = cbase + (size_t)t * CONV_DIM + D_INNER + D_STATE;
#pragma unroll
    for (int j = 0; j < 16; ++j) lCl[t][n0 + j] = (__bf16)(lam * rowp[n0 + j]);
    const int n = tid >> 2, p0 = (tid & 3) * 16;
#pragma unroll
    for (int j = 0; j < 16; ++j) lHT[p0 + j][n] = (__bf16)hbase[(size_t)n * HEADDIM + p0 + j];
  }
  __syncthreads();

  const int cn  = lane & 15;
  const int cmb = (lane >> 4) * 8;
  float* ybase = y + ((size_t)b * SEQ + (size_t)c * QL) * D_INNER + hh * HEADDIM;
#pragma unroll
  for (int rep = 0; rep < 2; ++rep) {
    const int ti = wave + rep * 8;
    const int tr = (ti >> 2) * 16, tc = (ti & 3) * 16;
    v8f acc;
#pragma unroll
    for (int v = 0; v < 8; ++v) acc[v] = 0.f;
#pragma unroll
    for (int kk = 0; kk < 2; ++kk)
      acc = __builtin_amdgcn_wmma_f32_16x16x32_bf16(
          false, afrag_load(&lCl[0][0], LDB, tr, kk * 32, lane),
          false, bfrag_load(&lHT[0][0], LDB, tc, kk * 32, lane),
          (short)0, acc, false, false);
#pragma unroll
    for (int v = 0; v < 8; ++v) {
      const int t = tr + cmb + v, p = tc + cn;
      ybase[(size_t)t * D_INNER + p] += acc[v];
    }
  }
}

// ---------------------------------------------------------------------------
// y = RMSNorm( y * SiLU(z) ) * norm_w, in place. One block / token.
// ---------------------------------------------------------------------------
__global__ __launch_bounds__(256) void gate_rms_kernel(
    float* __restrict__ y, const float* __restrict__ zx,
    const float* __restrict__ nw)
{
  const int t = blockIdx.x, tid = threadIdx.x;
  float* yrow = y + (size_t)t * D_INNER;
  const float* zrow = zx + (size_t)t * D_IN_PROJ;
  float v[4]; float ss = 0.f;
#pragma unroll
  for (int j = 0; j < 4; ++j) {
    const int e = tid + j * 256;
    const float z = zrow[e];
    const float g = yrow[e] * (z * sigmoidf_(z));
    v[j] = g; ss += g * g;
  }
  __shared__ float red[256];
  red[tid] = ss; __syncthreads();
  for (int s = 128; s > 0; s >>= 1) { if (tid < s) red[tid] += red[tid + s]; __syncthreads(); }
  const float rs = rsqrtf(red[0] * (1.f / 1024.f) + EPSV);
#pragma unroll
  for (int j = 0; j < 4; ++j) {
    const int e = tid + j * 256;
    yrow[e] = v[j] * rs * nw[e];
  }
}

// ---------------------------------------------------------------------------
// Classifier: out[t][c] = h[t] . Wc[c] + bc[c]   (N_CLASSES = 2)
// ---------------------------------------------------------------------------
__global__ __launch_bounds__(128) void cls_kernel(
    const float* __restrict__ h, const float* __restrict__ Wc,
    const float* __restrict__ bcv, float* __restrict__ out)
{
  const int t = blockIdx.x, tid = threadIdx.x;
  const float* row = h + (size_t)t * D_MODEL;
  float a0 = 0.f, a1 = 0.f;
  for (int k = tid; k < D_MODEL; k += 128) {
    const float v = row[k];
    a0 += v * Wc[k];
    a1 += v * Wc[D_MODEL + k];
  }
  __shared__ float r0[128], r1[128];
  r0[tid] = a0; r1[tid] = a1; __syncthreads();
  for (int s = 64; s > 0; s >>= 1) {
    if (tid < s) { r0[tid] += r0[tid + s]; r1[tid] += r1[tid + s]; }
    __syncthreads();
  }
  if (tid == 0) {
    out[(size_t)t * 2]     = r0[0] + bcv[0];
    out[(size_t)t * 2 + 1] = r1[0] + bcv[1];
  }
}

// ---------------------------------------------------------------------------
extern "C" void kernel_launch(void* const* d_in, const int* in_sizes, int n_in,
                              void* d_out, int out_size, void* d_ws, size_t ws_size,
                              hipStream_t stream)
{
  const float* x      = (const float*)d_in[0];
  const float* W1     = (const float*)d_in[1];
  const float* b1     = (const float*)d_in[2];
  const float* lnw    = (const float*)d_in[3];
  const float* lnb    = (const float*)d_in[4];
  const float* Win    = (const float*)d_in[5];
  const float* convw  = (const float*)d_in[6];
  const float* convb  = (const float*)d_in[7];
  const float* dtbias = (const float*)d_in[8];
  const float* Alog   = (const float*)d_in[9];
  const float* DskA   = (const float*)d_in[10];
  const float* normw  = (const float*)d_in[11];
  const float* Wout   = (const float*)d_in[12];
  const float* Wc     = (const float*)d_in[13];
  const float* bcv    = (const float*)d_in[14];
  float* out = (float*)d_out;

  float* ws = (float*)d_ws;
  size_t off = 0;
  float* h     = ws + off; off += (size_t)NTOK * D_MODEL;
  float* hln   = ws + off; off += (size_t)NTOK * D_MODEL;
  float* zx    = ws + off; off += (size_t)NTOK * D_IN_PROJ;
  float* cv    = ws + off; off += (size_t)NTOK * CONV_DIM;
  float* dtb   = ws + off; off += (size_t)NTOK * NHEADS;
  float* ybuf  = ws + off; off += (size_t)NTOK * D_INNER;
  float* Hbuf  = ws + off; off += (size_t)NBATCH * NHEADS * NCHUNK * D_STATE * HEADDIM;
  float* dAc   = ws + off; off += (size_t)NBATCH * NHEADS * NCHUNK;
  float* scumb = ws + off; off += (size_t)NBATCH * NHEADS * SEQ;

  const dim3 blk(256);
  const int ntile = NBATCH * NHEADS * NCHUNK;   // 4096 SSD tiles

  // h = GELU(x @ W1^T + b1)
  gemm_bf16_wmma<<<dim3(NTOK / TM, (D_MODEL + TN - 1) / TN), blk, 0, stream>>>(
      x, W1, b1, nullptr, h, NTOK, D_MODEL, D_IN, 1);

  for (int i = 0; i < NLAYERS; ++i) {
    layernorm_kernel<<<NTOK, blk, 0, stream>>>(h, lnw + i * D_MODEL, lnb + i * D_MODEL, hln);
    gemm_bf16_wmma<<<dim3(NTOK / TM, (D_IN_PROJ + TN - 1) / TN), blk, 0, stream>>>(
        hln, Win + (size_t)i * D_IN_PROJ * D_MODEL, nullptr, nullptr, zx,
        NTOK, D_IN_PROJ, D_MODEL, 0);
    convdt_kernel<<<NTOK, blk, 0, stream>>>(
        zx, convw + (size_t)i * CONV_DIM * D_CONV, convb + (size_t)i * CONV_DIM,
        dtbias + i * NHEADS, cv, dtb);
    ssd_pass1<<<ntile, blk, 0, stream>>>(
        cv, dtb, Alog + i * NHEADS, DskA + i * NHEADS, ybuf, Hbuf, dAc, scumb);
    ssd_pass2<<<NBATCH * NHEADS, blk, 0, stream>>>(Hbuf, dAc);
    ssd_pass3<<<ntile, blk, 0, stream>>>(Hbuf, cv, scumb, ybuf);
    gate_rms_kernel<<<NTOK, blk, 0, stream>>>(ybuf, zx, normw + (size_t)i * D_INNER);
    gemm_bf16_wmma<<<dim3(NTOK / TM, (D_MODEL + TN - 1) / TN), blk, 0, stream>>>(
        ybuf, Wout + (size_t)i * D_MODEL * D_INNER, nullptr, h, h,
        NTOK, D_MODEL, D_INNER, 0);
  }

  cls_kernel<<<NTOK, dim3(128), 0, stream>>>(h, Wc, bcv, out);

  (void)in_sizes; (void)n_in; (void)out_size; (void)ws_size;
}
